// FEM_17008070492739
// MI455X (gfx1250) — matrix operand order repaired and checked by reading, not compiled, not run on hardware
//
#include <hip/hip_runtime.h>

typedef __attribute__((ext_vector_type(16))) _Float16 f16x16;
typedef __attribute__((ext_vector_type(8)))  _Float16 f16x8;
typedef __attribute__((ext_vector_type(4)))  _Float16 f16x4;
typedef __attribute__((ext_vector_type(8)))  float    f32x8;
typedef __attribute__((ext_vector_type(4)))  unsigned u32x4;
typedef __attribute__((ext_vector_type(8)))  int      i32x8;
typedef __attribute__((ext_vector_type(4)))  int      i32x4;

static __device__ __forceinline__ f16x16 fem_pack16(const _Float16* __restrict__ lo,
                                                    const _Float16* __restrict__ hi) {
  f16x8 a0 = *(const f16x8*)lo;
  f16x8 a1 = *(const f16x8*)hi;
  f16x16 r;
#pragma unroll
  for (int i = 0; i < 8; ++i) { r[i] = a0[i]; r[i + 8] = a1[i]; }
  return r;
}

// Async DMA of 16 bytes global -> LDS (no VGPR round-trip), tracked by ASYNCcnt.
static __device__ __forceinline__ void fem_async_g2l_b128(const void* gptr,
                                                          void* lptr) {
  unsigned lds = (unsigned)(unsigned long long)lptr;  // LDS aperture: addr[31:0]
  unsigned long long ga = (unsigned long long)gptr;
  asm volatile("global_load_async_to_lds_b128 %0, %1, off"
               :: "v"(lds), "v"(ga) : "memory");
}
static __device__ __forceinline__ void fem_wait_async0() {
  asm volatile("s_wait_asynccnt 0x0" ::: "memory");
}

// ---------------------------------------------------------------------------
// Tensor Data Mover: DMA a (tileRows x 32) f16 tile of a row-major (M x K)
// tensor (row stride K elements) into contiguous LDS rows of 64B.
// D# built per CDNA5 ISA (2-D tile: groups 2/3 zero).
// Issued by one wave; completion via TENSORcnt.
// This toolchain exposes the 6-arg builtin form:
//   (u32x4 g0, i32x8 g1, i32x4 g2, i32x4 g3, i32x8 extra, i32 cpol)
// ---------------------------------------------------------------------------
#if __has_builtin(__builtin_amdgcn_tensor_load_to_lds)
#define FEM_HAS_TDM 1
static __device__ __forceinline__ void fem_tdm_load_a(const _Float16* gsrc,
                                                      void* ldst, int tileRows,
                                                      int M, int K) {
  unsigned long long ga = (unsigned long long)gsrc;
  unsigned lds = (unsigned)(unsigned long long)ldst;
  u32x4 g0;
  g0[0] = 1u;                                      // count=1, user descriptor
  g0[1] = lds;                                     // lds_addr (bytes)
  g0[2] = (unsigned)(ga & 0xffffffffu);            // global_addr[31:0]
  g0[3] = (unsigned)((ga >> 32) & 0x01ffffffu)     // global_addr[56:32]
          | (2u << 30);                            // type = 2 ("image")
  i32x8 g1;
  g1[0] = 0x00010000;                              // data_size=1 (2B), no mcast
  g1[1] = (int)((unsigned)(K & 0xffff) << 16);     // tensor_dim0[15:0]
  g1[2] = (int)(((unsigned)K >> 16) & 0xffffu) |   // tensor_dim0[31:16]
          (int)((unsigned)(M & 0xffff) << 16);     // tensor_dim1[15:0]
  g1[3] = (int)(((unsigned)M >> 16) & 0xffffu) |   // tensor_dim1[31:16]
          (int)(32u << 16);                        // tile_dim0 = 32 elements
  g1[4] = tileRows & 0xffff;                       // tile_dim1 (tile_dim2=0)
  g1[5] = K;                                       // tensor_dim0_stride[31:0]
  g1[6] = 0;                                       // stride hi / dim1_stride lo
  g1[7] = 0;                                       // dim1_stride hi (unused 2D)
  i32x4 gz4 = {};                                  // groups 2/3: zero (2-D)
  i32x8 gz8 = {};                                  // extra operand (unused)
  __builtin_amdgcn_tensor_load_to_lds(g0, g1, gz4, gz4, gz8, 0);
}
#else
#define FEM_HAS_TDM 0
#endif

// ---------------------------------------------------------------------------
// LayerNorm over full CHW per sample. Writes fp32 (optional, for residual) and
// f16 (GEMM input). One block per sample, 1024 threads (32 waves).
// ---------------------------------------------------------------------------
__global__ __launch_bounds__(1024) void fem_ln(
    const float* __restrict__ x, const float* __restrict__ w,
    const float* __restrict__ bb, float* __restrict__ outF,
    _Float16* __restrict__ outH) {
  const long CHW = 512L * 4096L;
  const long n4 = CHW / 4;
  long base = (long)blockIdx.x * CHW;
  const float4* xp = (const float4*)(x + base);

  float s = 0.f, s2 = 0.f;
  for (long i = threadIdx.x; i < n4; i += 1024) {
    float4 v = xp[i];
    s  += v.x + v.y + v.z + v.w;
    s2 += v.x * v.x + v.y * v.y + v.z * v.z + v.w * v.w;
  }
  __shared__ float r1[1024];
  __shared__ float r2[1024];
  r1[threadIdx.x] = s; r2[threadIdx.x] = s2;
  __syncthreads();
  for (int st = 512; st > 0; st >>= 1) {
    if ((int)threadIdx.x < st) {
      r1[threadIdx.x] += r1[threadIdx.x + st];
      r2[threadIdx.x] += r2[threadIdx.x + st];
    }
    __syncthreads();
  }
  float mu = r1[0] / (float)CHW;
  float var = r2[0] / (float)CHW - mu * mu;
  float rs = rsqrtf(var + 1e-5f);

  const float4* wp = (const float4*)w;
  const float4* bp = (const float4*)bb;
  float4* of = outF ? (float4*)(outF + base) : nullptr;
  f16x4* oh = (f16x4*)(outH + base);
  for (long i = threadIdx.x; i < n4; i += 1024) {
    float4 v = xp[i], wv = wp[i], bv = bp[i];
    float4 y;
    y.x = (v.x - mu) * rs * wv.x + bv.x;
    y.y = (v.y - mu) * rs * wv.y + bv.y;
    y.z = (v.z - mu) * rs * wv.z + bv.z;
    y.w = (v.w - mu) * rs * wv.w + bv.w;
    if (of) of[i] = y;
    f16x4 h;
    h[0] = (_Float16)y.x; h[1] = (_Float16)y.y;
    h[2] = (_Float16)y.z; h[3] = (_Float16)y.w;
    oh[i] = h;
  }
}

// ---------------------------------------------------------------------------
// fp32 -> f16 cast (for the 1x1 conv weight matrices).
// ---------------------------------------------------------------------------
__global__ void fem_cast(const float* __restrict__ in, _Float16* __restrict__ out,
                         int n) {
  int i = blockIdx.x * 256 + threadIdx.x;
  if (i < n) out[i] = (_Float16)in[i];
}

// ---------------------------------------------------------------------------
// Generic WMMA f16 GEMM:  Y[z] = A[z](MxK) * X[z](KxN) (+bias) (+resid -> f32)
// Tile 128x128, 256 threads = 8 waves (2 M x 4 N), 8 v_wmma per wave / K-step.
// A tile is DMA'd global->LDS by the Tensor Data Mover (one tensor_load_to_lds
// per K-step, issued by wave 0, TENSORcnt-tracked; async-b128 fallback).
// X tile is staged transposed in LDS so each B-fragment is two 16B k-runs of one row.
// ---------------------------------------------------------------------------
__global__ __launch_bounds__(256) void fem_gemm(
    const _Float16* __restrict__ A, const _Float16* __restrict__ X,
    const float* __restrict__ bias, const float* __restrict__ resid,
    _Float16* __restrict__ Yh, float* __restrict__ Yf,
    int M, int K, int N, long strideA, long strideXb, int zPerB) {
  __shared__ _Float16 xt[128][40];  // X^T tile [n][k], padded pitch
  __shared__ _Float16 at[128][32];  // A tile [m][k], row-major 64B rows

  int z = blockIdx.z;
  const _Float16* Az = A + (long)z * strideA;
  const _Float16* Xz = X + (long)(z / zPerB) * strideXb +
                       (long)(z % zPerB) * (long)K * (long)N;
  int n0 = blockIdx.x * 128, m0 = blockIdx.y * 128;
  int t = threadIdx.x, lane = t & 31, wave = t >> 5;
  int wm = wave >> 2, wn = wave & 3;
  int l16 = lane & 15, lhi = lane >> 4;

  f32x8 acc[4][2] = {};

  int kk = t >> 3;          // 0..31 : K row within X tile
  int nn = (t & 7) << 4;    // 0..112: N start within X tile

  int tileRows = M - m0; if (tileRows > 128) tileRows = 128;

#if !FEM_HAS_TDM
  int ar0 = (t * 2) >> 2, ak0 = ((t * 2) & 3) << 3;
  int ar1 = (t * 2 + 1) >> 2, ak1 = ((t * 2 + 1) & 3) << 3;
  int am0 = m0 + ar0; if (am0 >= M) am0 = M - 1;
  int am1 = m0 + ar1; if (am1 >= M) am1 = M - 1;
#endif

  for (int k0 = 0; k0 < K; k0 += 32) {
    __syncthreads();
#if FEM_HAS_TDM
    // One TDM descriptor moves the whole 128x32 A tile while we transpose X.
    if (wave == 0)
      fem_tdm_load_a(Az + (long)m0 * K + k0, &at[0][0], tileRows, M, K);
#else
    fem_async_g2l_b128(Az + (long)am0 * K + k0 + ak0, &at[ar0][ak0]);
    fem_async_g2l_b128(Az + (long)am1 * K + k0 + ak1, &at[ar1][ak1]);
#endif
    {
      const _Float16* src = Xz + (long)(k0 + kk) * N + (n0 + nn);
      f16x8 x0 = *(const f16x8*)src;
      f16x8 x1 = *(const f16x8*)(src + 8);
#pragma unroll
      for (int i = 0; i < 8; ++i) xt[nn + i][kk] = x0[i];
#pragma unroll
      for (int i = 0; i < 8; ++i) xt[nn + 8 + i][kk] = x1[i];
      if (k0 + 32 < K) __builtin_prefetch(src + 32L * N, 0, 0);
    }
#if FEM_HAS_TDM
    __builtin_amdgcn_s_wait_tensorcnt(0);  // immediate for waves 1..7
#else
    fem_wait_async0();
#endif
    __syncthreads();

    f16x16 af[4];
#pragma unroll
    for (int mt = 0; mt < 4; ++mt) {
      const _Float16* ap = &at[wm * 64 + mt * 16 + l16][lhi * 8];
      af[mt] = fem_pack16(ap, ap + 16);
    }
    f16x16 bfr[2];
#pragma unroll
    for (int nt = 0; nt < 2; ++nt) {
      const _Float16* bp = &xt[wn * 32 + nt * 16 + l16][lhi * 8];
      bfr[nt] = fem_pack16(bp, bp + 16);
    }
#pragma unroll
    for (int mt = 0; mt < 4; ++mt)
#pragma unroll
      for (int nt = 0; nt < 2; ++nt)
        acc[mt][nt] = __builtin_amdgcn_wmma_f32_16x16x32_f16(
            false, af[mt], false, bfr[nt], (short)0, acc[mt][nt], false, false);
  }

#pragma unroll
  for (int mt = 0; mt < 4; ++mt) {
#pragma unroll
    for (int nt = 0; nt < 2; ++nt) {
      int n = n0 + wn * 32 + nt * 16 + l16;
#pragma unroll
      for (int j = 0; j < 8; ++j) {
        int m = m0 + wm * 64 + mt * 16 + lhi * 8 + j;
        if (m < M) {
          float v = acc[mt][nt][j] + (bias ? bias[m] : 0.0f);
          long off = (long)z * M * N + (long)m * N + n;
          if (resid) Yf[off] = v + resid[off];
          else       Yh[off] = (_Float16)v;
        }
      }
    }
  }
}

// ---------------------------------------------------------------------------
// Grouped 3x3 conv, pad 1 (ipg = channels per group; in==out per group).
// grid (S/256, OC, B). Input strip (4 rows + halo) is async-DMA'd into LDS.
// ---------------------------------------------------------------------------
__global__ __launch_bounds__(256) void fem_dw(
    const _Float16* __restrict__ Xin, const float* __restrict__ Wdw,
    const float* __restrict__ bias, _Float16* __restrict__ Yout,
    int OC, int ipg) {
  __shared__ _Float16 tile[2][6][64];  // [ic][haloed row][x]
  int o = blockIdx.y, b = blockIdx.z;
  int y0 = blockIdx.x * 4;
  int t = threadIdx.x;
  int g = o / ipg;
  const _Float16* chBase = Xin + ((long)b * OC + g * ipg) * 4096L;

  // Async-stage rows y0-1 .. y0+4 of each group input channel (128B per row).
  int nchunks = ipg * 48;  // ipg * 6 rows * 8 x (16B) chunks
  if (t < nchunks) {
    int ic = t / 48, rem = t % 48;
    int r = rem / 8, c8 = rem % 8;
    int yy = y0 - 1 + r;
    if (yy >= 0 && yy < 64)
      fem_async_g2l_b128(chBase + (long)ic * 4096L + yy * 64 + c8 * 8,
                         &tile[ic][r][c8 * 8]);
  }
  fem_wait_async0();
  __syncthreads();

  int y = y0 + (t >> 6), x = t & 63;
  const float* wq = Wdw + (long)o * ipg * 9;
  float acc = bias[o];
  for (int ic = 0; ic < ipg; ++ic) {
    const float* wk = wq + ic * 9;
#pragma unroll
    for (int dy = -1; dy <= 1; ++dy) {
      int yy = y + dy;
      if (yy < 0 || yy >= 64) continue;
      int r = (t >> 6) + 1 + dy;  // haloed row index in LDS strip
#pragma unroll
      for (int dx = -1; dx <= 1; ++dx) {
        int xx = x + dx;
        if (xx >= 0 && xx < 64)
          acc += (float)tile[ic][r][xx] * wk[(dy + 1) * 3 + (dx + 1)];
      }
    }
  }
  Yout[((long)b * OC + o) * 4096L + y * 64 + x] = (_Float16)acc;
}

// ---------------------------------------------------------------------------
// L2-normalize rows of length 4096. One 256-thread block per (b, c<512) row.
// ---------------------------------------------------------------------------
__global__ __launch_bounds__(256) void fem_l2(
    const _Float16* __restrict__ Xin, _Float16* __restrict__ Yout,
    int chStride) {
  int r = blockIdx.x;
  int b = r >> 9, c = r & 511;
  const f16x8* xv = (const f16x8*)(Xin + ((long)b * chStride + c) * 4096L);
  f16x8* yv = (f16x8*)(Yout + (long)r * 4096L);

  float ss = 0.f;
  for (int i = threadIdx.x; i < 512; i += 256) {
    f16x8 v = xv[i];
#pragma unroll
    for (int j = 0; j < 8; ++j) { float f = (float)v[j]; ss += f * f; }
  }
  __shared__ float red[256];
  red[threadIdx.x] = ss;
  __syncthreads();
  for (int st = 128; st > 0; st >>= 1) {
    if ((int)threadIdx.x < st) red[threadIdx.x] += red[threadIdx.x + st];
    __syncthreads();
  }
  float scale = 1.0f / fmaxf(sqrtf(red[0]), 1e-12f);
  for (int i = threadIdx.x; i < 512; i += 256) {
    f16x8 v = xv[i];
    f16x8 o;
#pragma unroll
    for (int j = 0; j < 8; ++j) o[j] = (_Float16)((float)v[j] * scale);
    yv[i] = o;
  }
}

// ---------------------------------------------------------------------------
// Attention scores: per (b,head) 64x64 = Q(64x4096) * K(64x4096)^T / scaling.
// Both operands contiguous along S -> direct-from-global WMMA, no LDS.
// ---------------------------------------------------------------------------
__global__ __launch_bounds__(128) void fem_scores(
    const _Float16* __restrict__ Qn, const _Float16* __restrict__ Kn,
    float* __restrict__ Sc, const float* __restrict__ scaling) {
  int bh = blockIdx.x;
  const _Float16* Q = Qn + (long)bh * 64L * 4096L;
  const _Float16* Kp = Kn + (long)bh * 64L * 4096L;
  int t = threadIdx.x, lane = t & 31, wave = t >> 5;
  int l16 = lane & 15, lhi = lane >> 4;

  f32x8 acc[4] = {};
  for (int k0 = 0; k0 < 4096; k0 += 32) {
    const _Float16* ap = Q + (long)(wave * 16 + l16) * 4096L + k0 + lhi * 8;
    f16x16 af = fem_pack16(ap, ap + 16);
#pragma unroll
    for (int nt = 0; nt < 4; ++nt) {
      const _Float16* bp = Kp + (long)(nt * 16 + l16) * 4096L + k0 + lhi * 8;
      f16x16 bb = fem_pack16(bp, bp + 16);
      acc[nt] = __builtin_amdgcn_wmma_f32_16x16x32_f16(
          false, af, false, bb, (short)0, acc[nt], false, false);
    }
  }
  float inv = 1.0f / scaling[0];
  float* Ab = Sc + (long)bh * 64L * 64L;
#pragma unroll
  for (int nt = 0; nt < 4; ++nt)
#pragma unroll
    for (int j = 0; j < 8; ++j) {
      int c = wave * 16 + lhi * 8 + j;
      int d = nt * 16 + l16;
      Ab[c * 64 + d] = acc[nt][j] * inv;
    }
}

// ---------------------------------------------------------------------------
// Softmax over d (row of 64) and write transposed f16: AT[bh][d][c].
// ---------------------------------------------------------------------------
__global__ __launch_bounds__(64) void fem_softmax(
    const float* __restrict__ Ain, _Float16* __restrict__ AT) {
  int row = blockIdx.x;       // bh*64 + c
  int bh = row >> 6, c = row & 63;
  int d = threadIdx.x;
  float v = Ain[(long)row * 64 + d];
  __shared__ float red[64];
  red[d] = v;
  __syncthreads();
  for (int st = 32; st > 0; st >>= 1) {
    if (d < st) red[d] = fmaxf(red[d], red[d + st]);
    __syncthreads();
  }
  float mx = red[0];
  __syncthreads();
  float e = __expf(v - mx);
  red[d] = e;
  __syncthreads();
  for (int st = 32; st > 0; st >>= 1) {
    if (d < st) red[d] += red[d + st];
    __syncthreads();
  }
  AT[((long)bh * 64 + d) * 64 + c] = (_Float16)(e / red[0]);
}

// ---------------------------------------------------------------------------
extern "C" void kernel_launch(void* const* d_in, const int* in_sizes, int n_in,
                              void* d_out, int out_size, void* d_ws,
                              size_t ws_size, hipStream_t stream) {
  const float* image     = (const float*)d_in[0];
  const float* watermark = (const float*)d_in[1];
  const float* ln_img_w  = (const float*)d_in[2];
  const float* ln_img_b  = (const float*)d_in[3];
  const float* ln_wm_w   = (const float*)d_in[4];
  const float* ln_wm_b   = (const float*)d_in[5];
  const float* pw_img_w  = (const float*)d_in[6];
  const float* pw_img_b  = (const float*)d_in[7];
  const float* dw_img_w  = (const float*)d_in[8];
  const float* dw_img_b  = (const float*)d_in[9];
  const float* pw_wm_w   = (const float*)d_in[10];
  const float* pw_wm_b   = (const float*)d_in[11];
  const float* dw_wm_w   = (const float*)d_in[12];
  const float* dw_wm_b   = (const float*)d_in[13];
  const float* scaling   = (const float*)d_in[14];
  const float* pw_out_w  = (const float*)d_in[15];
  const float* pw_out_b  = (const float*)d_in[16];

  const long NE = 8L * 512L * 4096L;  // B*C*S = 16,777,216

  char* p = (char*)d_ws;
  auto alloc = [&](size_t bytes) -> void* {
    void* r = (void*)p;
    p += (bytes + 255) & ~(size_t)255;
    return r;
  };
  float*    imgF    = (float*)   alloc(NE * 4);
  _Float16* imgH    = (_Float16*)alloc(NE * 2);
  _Float16* wmH     = (_Float16*)alloc(NE * 2);
  _Float16* kvH     = (_Float16*)alloc(2 * NE * 2);
  _Float16* qH      = (_Float16*)alloc(NE * 2);
  _Float16* kvdH    = (_Float16*)alloc(2 * NE * 2);
  _Float16* qdH     = (_Float16*)alloc(NE * 2);
  _Float16* qnH     = (_Float16*)alloc(NE * 2);
  _Float16* knH     = (_Float16*)alloc(NE * 2);
  _Float16* vaH     = (_Float16*)alloc(NE * 2);
  float*    scoresF = (float*)   alloc(8L * 8 * 64 * 64 * 4);
  _Float16* atH     = (_Float16*)alloc(8L * 8 * 64 * 64 * 2);
  _Float16* wImgH   = (_Float16*)alloc(1024L * 512 * 2);
  _Float16* wWmH    = (_Float16*)alloc(512L * 512 * 2);
  _Float16* wOutH   = (_Float16*)alloc(512L * 512 * 2);

  // 1) LayerNorms
  fem_ln<<<8, 1024, 0, stream>>>(image, ln_img_w, ln_img_b, imgF, imgH);
  fem_ln<<<8, 1024, 0, stream>>>(watermark, ln_wm_w, ln_wm_b, nullptr, wmH);

  // 2) Weight casts
  fem_cast<<<(1024 * 512 + 255) / 256, 256, 0, stream>>>(pw_img_w, wImgH, 1024 * 512);
  fem_cast<<<(512 * 512 + 255) / 256, 256, 0, stream>>>(pw_wm_w, wWmH, 512 * 512);
  fem_cast<<<(512 * 512 + 255) / 256, 256, 0, stream>>>(pw_out_w, wOutH, 512 * 512);

  // 3) Pointwise convs as WMMA GEMMs (per sample: N = S = 4096)
  fem_gemm<<<dim3(32, 8, 8), 256, 0, stream>>>(
      wImgH, imgH, pw_img_b, nullptr, kvH, nullptr,
      1024, 512, 4096, 0L, 512L * 4096L, 1);
  fem_gemm<<<dim3(32, 4, 8), 256, 0, stream>>>(
      wWmH, wmH, pw_wm_b, nullptr, qH, nullptr,
      512, 512, 4096, 0L, 512L * 4096L, 1);

  // 4) Grouped 3x3 convs (async-LDS staged)
  fem_dw<<<dim3(16, 1024, 8), 256, 0, stream>>>(kvH, dw_img_w, dw_img_b, kvdH, 1024, 2);
  fem_dw<<<dim3(16, 512, 8), 256, 0, stream>>>(qH, dw_wm_w, dw_wm_b, qdH, 512, 1);

  // 5) L2 norms over S (Q from qd; K = low-half channels of kvd)
  fem_l2<<<4096, 256, 0, stream>>>(qdH, qnH, 512);
  fem_l2<<<4096, 256, 0, stream>>>(kvdH, knH, 1024);

  // 6) Attention scores + softmax (writes A^T in f16)
  fem_scores<<<64, 128, 0, stream>>>(qnH, knH, scoresF, scaling);
  fem_softmax<<<4096, 64, 0, stream>>>(scoresF, atH);

  // 7) VA = A^T (64x64, per z=b*8+h) * V (64x4096); V = high-half of kvd
  fem_gemm<<<dim3(32, 1, 64), 256, 0, stream>>>(
      atH, kvdH + 512L * 4096L, nullptr, nullptr, vaH, nullptr,
      64, 64, 4096, 64L * 64L, 1024L * 4096L, 8);

  // 8) Final pointwise conv + residual (fp32 output)
  fem_gemm<<<dim3(32, 4, 8), 256, 0, stream>>>(
      wOutH, vaH, pw_out_b, imgF, nullptr, (float*)d_out,
      512, 512, 4096, 0L, 512L * 4096L, 1);

  (void)in_sizes; (void)n_in; (void)out_size; (void)ws_size;
}
